// LigColumnAttention_2199023255607
// MI455X (gfx1250) — compile-verified, matrix operand order checked by reading, not run on hardware
//
#include <hip/hip_runtime.h>
#include <math.h>

// ---------------------------------------------------------------------------
// LigColumnAttention for MI455X (gfx1250), wave32 + v_wmma_f32_16x16x32_f16,
// LDS staging via global_load_async_to_lds_b128 (ASYNCcnt path).
// B=1, M=128, N=256, C=256, heads=8, head_dim=32.
// ---------------------------------------------------------------------------

#define M_   128
#define N_   256
#define C_   256
#define AC   32
#define H_   8
#define EPS_ 1e-5f

typedef _Float16 half_t;
typedef _Float16 v16h __attribute__((ext_vector_type(16)));
typedef _Float16 v8h  __attribute__((ext_vector_type(8)));
typedef float    v8f  __attribute__((ext_vector_type(8)));

// ---- WMMA helpers ----------------------------------------------------------
__device__ __forceinline__ v8f vzero8() {
  v8f z;
#pragma unroll
  for (int i = 0; i < 8; ++i) z[i] = 0.0f;
  return z;
}

__device__ __forceinline__ v8f wmma16(v16h a, v16h b, v8f c) {
  // D = A(16x32 f16) * B(32x16 f16) + C(16x16 f32)
  return __builtin_amdgcn_wmma_f32_16x16x32_f16(
      false, a, false, b, (short)0, c, false, false);
}

// A/B fragment loader. ISA 7.12.2 (16-bit A 16x32): per lane (g = lane>>4),
// halves 0..7  <- k_base + g*8 .. +7   (contiguous)
// halves 8..15 <- k_base + 16 + g*8 .. (contiguous)
// => two 16-byte DS loads per fragment.
__device__ __forceinline__ v16h load_frag(const half_t* row, int kb, int g) {
  v16h r;
  *(v8h*)&r       = *(const v8h*)(row + kb + g * 8);
  *((v8h*)&r + 1) = *(const v8h*)(row + kb + 16 + g * 8);
  return r;
}

// ---- CDNA5 async global->LDS copy (16B per lane), tracked by ASYNCcnt ------
__device__ __forceinline__ void cp_async16(void* lds, const void* gsrc) {
  asm volatile("global_load_async_to_lds_b128 %0, %1, off"
               :: "v"((unsigned)(unsigned long long)lds), "v"(gsrc)
               : "memory");
}
__device__ __forceinline__ void cp_async_wait() {
  asm volatile("s_wait_asynccnt 0" ::: "memory");
}

// ---------------------------------------------------------------------------
// Kernel 1: LayerNorm over C=256, emit f16. One wave per row.
// ---------------------------------------------------------------------------
__global__ __launch_bounds__(256) void ln_kernel(
    const float* __restrict__ x, const float* __restrict__ gamma,
    const float* __restrict__ beta, half_t* __restrict__ xh) {
  const int wave = threadIdx.x >> 5;
  const int lane = threadIdx.x & 31;
  const long row = (long)blockIdx.x * 8 + wave;  // 32768 rows total
  const float* xr = x + row * C_;
  const int c0 = lane * 8;

  float v[8];
  float s = 0.0f, ss = 0.0f;
#pragma unroll
  for (int i = 0; i < 8; ++i) {
    v[i] = xr[c0 + i];
    s += v[i];
    ss += v[i] * v[i];
  }
#pragma unroll
  for (int off = 16; off > 0; off >>= 1) {
    s  += __shfl_xor(s, off, 32);
    ss += __shfl_xor(ss, off, 32);
  }
  const float mean = s * (1.0f / 256.0f);
  const float var  = ss * (1.0f / 256.0f) - mean * mean;
  const float rstd = rsqrtf(var + EPS_);

  v8h o;
#pragma unroll
  for (int i = 0; i < 8; ++i)
    o[i] = (half_t)((v[i] - mean) * rstd * gamma[c0 + i] + beta[c0 + i]);
  *(v8h*)(xh + row * C_ + c0) = o;
}

// ---------------------------------------------------------------------------
// Kernel 2: transpose + f16-convert the four 256x256 weights.
// wT[n*256 + k] = (f16) w[k*256 + n]  (so B-fragments load contiguously)
// ---------------------------------------------------------------------------
__global__ __launch_bounds__(256) void wcvt_kernel(
    const float* __restrict__ wq, const float* __restrict__ wk,
    const float* __restrict__ wv, const float* __restrict__ wf,
    half_t* __restrict__ wqT, half_t* __restrict__ wkT,
    half_t* __restrict__ wvT, half_t* __restrict__ wfT) {
  const float* src;
  half_t* dst;
  switch (blockIdx.y) {
    case 0:  src = wq; dst = wqT; break;
    case 1:  src = wk; dst = wkT; break;
    case 2:  src = wv; dst = wvT; break;
    default: src = wf; dst = wfT; break;
  }
  const int idx = blockIdx.x * 256 + threadIdx.x;  // 0..65535
  const int n = idx >> 8, k = idx & 255;
  dst[idx] = (half_t)src[k * 256 + n];
}

// ---------------------------------------------------------------------------
// Kernel 3: QKV projection GEMM [32768,256] x [256,256] (f16 in, f16 out).
// LDS: full W^T at offset 0 (so all 16 n-tile bases fit the DS immediate),
// then X block.  Wave w owns i-tile w x 16 n-tiles; inner loop is software-
// pipelined (B for nt+1 / A for kt+1 loaded before the WMMA for nt).
// Q/K written [m][h][t][c]; V written transposed [m][h][c][t].
// ---------------------------------------------------------------------------
__global__ __launch_bounds__(256) void qkv_kernel(
    const half_t* __restrict__ xh, const half_t* __restrict__ wT,
    half_t* __restrict__ dst, int is_v) {
  extern __shared__ char smem[];
  half_t* Ws = (half_t*)smem;                       // [256 n][256 k]  (128 KB)
  half_t* Xs = (half_t*)(smem + 256 * 256 * 2);     // [128][256]      (64 KB)

  const int tid = threadIdx.x;
  const long rowBase = (long)blockIdx.x * 128;
  {
    const float4* gw = (const float4*)wT;
    float4* sw = (float4*)Ws;
    for (int i = tid; i < 256 * 256 / 8; i += 256) cp_async16(sw + i, gw + i);
    const float4* gx = (const float4*)(xh + rowBase * C_);
    float4* sx = (float4*)Xs;
    for (int i = tid; i < 128 * 256 / 8; i += 256) cp_async16(sx + i, gx + i);
  }
  cp_async_wait();
  __syncthreads();

  const int wave = tid >> 5, lane = tid & 31;
  const int ln = lane & 15, g = lane >> 4;

  v8f acc[16];
#pragma unroll
  for (int nt = 0; nt < 16; ++nt) acc[nt] = vzero8();

  const half_t* arow = Xs + (wave * 16 + ln) * 256;
  const half_t* brow = Ws + ln * 256;  // + nt*4096 halves (8 KB DS immediates)

  v16h a = load_frag(arow, 0, g);
  v16h b = load_frag(brow, 0, g);
#pragma unroll
  for (int kt = 0; kt < 8; ++kt) {
    const int kb = kt * 32;
    v16h a_next = a, b_next = b;
#pragma unroll
    for (int nt = 0; nt < 16; ++nt) {
      if (nt < 15) {
        b_next = load_frag(brow + (nt + 1) * 4096, kb, g);
      } else if (kt < 7) {
        a_next = load_frag(arow, kb + 32, g);
        b_next = load_frag(brow, kb + 32, g);
      }
      acc[nt] = wmma16(a, b, acc[nt]);
      b = b_next;
    }
    a = a_next;
  }

#pragma unroll
  for (int nt = 0; nt < 16; ++nt) {
    const int col = nt * 16 + ln;
    const int h = col >> 5, c = col & 31;
#pragma unroll
    for (int r = 0; r < 8; ++r) {
      const int row = wave * 16 + g * 8 + r;
      const long tG = rowBase + row;
      const int m = (int)(tG >> 8);
      const int t = (int)(tG & 255);
      const half_t val = (half_t)acc[nt][r];
      if (is_v)
        dst[(((long)m * H_ + h) * AC + c) * N_ + t] = val;   // V^T: [m][h][c][t]
      else
        dst[(((long)m * H_ + h) * N_ + t) * AC + c] = val;   // Q/K: [m][h][t][c]
    }
  }
}

// ---------------------------------------------------------------------------
// Kernel 4: attention per (m,h). LDS: Qh/Kh [256][32], V^T [32][256] (48KB)
// + S [256][256] f32 (256KB) = 304KB (<320KB WGP LDS).
// S=Q*K^T is ONE wmma per 16x16 tile (K=head_dim=32). Softmax thread-per-row
// (scale folded into exp2), P overwrites S in-place as f16, then O = P*V.
// ---------------------------------------------------------------------------
__global__ __launch_bounds__(256) void attn_kernel(
    const half_t* __restrict__ Q, const half_t* __restrict__ K,
    const half_t* __restrict__ Vt, half_t* __restrict__ O) {
  extern __shared__ char smem[];
  half_t* Qs = (half_t*)smem;            // [256][32]
  half_t* Ks = Qs + 256 * 32;            // [256][32]
  half_t* Vs = Ks + 256 * 32;            // [32][256]
  float*  S  = (float*)(Vs + 32 * 256);  // [256][256] f32, rows reused as P f16

  const int m = blockIdx.x, h = blockIdx.y;
  const int tid = threadIdx.x;
  const long base_qk = ((long)m * H_ + h) * N_ * AC;
  const long base_v  = ((long)m * H_ + h) * AC * N_;
  {
    const float4* gq = (const float4*)(Q + base_qk);
    const float4* gk = (const float4*)(K + base_qk);
    const float4* gv = (const float4*)(Vt + base_v);
    float4* sq = (float4*)Qs;
    float4* sk = (float4*)Ks;
    float4* sv = (float4*)Vs;
    for (int i = tid; i < 1024; i += 256) {
      cp_async16(sq + i, gq + i);
      cp_async16(sk + i, gk + i);
      cp_async16(sv + i, gv + i);
    }
  }
  cp_async_wait();
  __syncthreads();

  const int wave = tid >> 5, lane = tid & 31;
  const int ln = lane & 15, g = lane >> 4;

  // ---- S = Q * K^T (f32), pipelined over j-tiles -------------------------
#pragma unroll
  for (int ii = 0; ii < 2; ++ii) {
    const int it = wave * 2 + ii;  // 16 i-tiles over 8 waves
    v16h a = load_frag(Qs + (it * 16 + ln) * 32, 0, g);
    v16h b = load_frag(Ks + ln * 32, 0, g);
#pragma unroll
    for (int jt = 0; jt < 16; ++jt) {
      v16h b_next = b;
      if (jt < 15) b_next = load_frag(Ks + ((jt + 1) * 16 + ln) * 32, 0, g);
      v8f d = wmma16(a, b, vzero8());
      b = b_next;
      float* srow = S + (it * 16 + g * 8) * 256 + jt * 16 + ln;
#pragma unroll
      for (int r = 0; r < 8; ++r) srow[r * 256] = d[r];
    }
  }
  __syncthreads();

  // ---- softmax over j, one thread per row; fold 1/sqrt(32) into exp2 -----
  {
    const int row = tid;
    float* srow = S + row * 256;
    const int rot = (row * 37) & 255;  // spread LDS banks across lanes
    float mx = -1e30f;
    for (int j = 0; j < 256; ++j) {
      const int jj = (j + rot) & 255;
      mx = fmaxf(mx, srow[jj]);
    }
    const float sc = 0.17677669529663687f * 1.4426950408889634f;  // scale*log2e
    float sum = 0.0f;
    for (int j = 0; j < 256; ++j) {
      const int jj = (j + rot) & 255;
      const float e = exp2f((srow[jj] - mx) * sc);
      srow[jj] = e;
      sum += e;
    }
    const float inv = 1.0f / sum;
    half_t* prow = (half_t*)((char*)S + row * 1024);
    // ascending in-place f32->f16: prow[j] clobbers bytes of srow[j>>1],
    // which has already been consumed.
    for (int j = 0; j < 256; ++j) prow[j] = (half_t)(srow[j] * inv);
  }
  __syncthreads();

  // ---- O = P * V  ([256,256] f16 x [256,32] via V^T), pipelined ----------
#pragma unroll
  for (int ii = 0; ii < 2; ++ii) {
    const int it = wave * 2 + ii;
    const half_t* prow = (const half_t*)((const char*)S + (it * 16 + ln) * 1024);
    v8f acc0 = vzero8(), acc1 = vzero8();
    v16h a  = load_frag(prow, 0, g);
    v16h b0 = load_frag(Vs + ln * 256, 0, g);
    v16h b1 = load_frag(Vs + (16 + ln) * 256, 0, g);
#pragma unroll
    for (int kt = 0; kt < 8; ++kt) {
      v16h an = a, b0n = b0, b1n = b1;
      if (kt < 7) {
        const int kb = kt * 32 + 32;
        an  = load_frag(prow, kb, g);
        b0n = load_frag(Vs + ln * 256, kb, g);
        b1n = load_frag(Vs + (16 + ln) * 256, kb, g);
      }
      acc0 = wmma16(a, b0, acc0);
      acc1 = wmma16(a, b1, acc1);
      a = an; b0 = b0n; b1 = b1n;
    }
#pragma unroll
    for (int r = 0; r < 8; ++r) {
      const int t = it * 16 + g * 8 + r;
      const long orow = ((long)m * N_ + t) * C_ + h * AC;  // O: [t][h*32+c]
      O[orow + ln]      = (half_t)acc0[r];
      O[orow + 16 + ln] = (half_t)acc1[r];
    }
  }
}

// ---------------------------------------------------------------------------
// Kernel 5: output projection Y = O @ wf + bf, fp32 out. Same tiling as QKV.
// ---------------------------------------------------------------------------
__global__ __launch_bounds__(256) void outproj_kernel(
    const half_t* __restrict__ Oh, const half_t* __restrict__ wfT,
    const float* __restrict__ bf, float* __restrict__ out) {
  extern __shared__ char smem[];
  half_t* Ws = (half_t*)smem;                    // [256][256] (128 KB)
  half_t* Xs = (half_t*)(smem + 256 * 256 * 2);  // [128][256] (64 KB)

  const int tid = threadIdx.x;
  const long rowBase = (long)blockIdx.x * 128;
  {
    const float4* gw = (const float4*)wfT;
    float4* sw = (float4*)Ws;
    for (int i = tid; i < 256 * 256 / 8; i += 256) cp_async16(sw + i, gw + i);
    const float4* gx = (const float4*)(Oh + rowBase * C_);
    float4* sx = (float4*)Xs;
    for (int i = tid; i < 128 * 256 / 8; i += 256) cp_async16(sx + i, gx + i);
  }
  cp_async_wait();
  __syncthreads();

  const int wave = tid >> 5, lane = tid & 31;
  const int ln = lane & 15, g = lane >> 4;

  v8f acc[16];
#pragma unroll
  for (int nt = 0; nt < 16; ++nt) acc[nt] = vzero8();

  const half_t* arow = Xs + (wave * 16 + ln) * 256;
  const half_t* brow = Ws + ln * 256;

  v16h a = load_frag(arow, 0, g);
  v16h b = load_frag(brow, 0, g);
#pragma unroll
  for (int kt = 0; kt < 8; ++kt) {
    const int kb = kt * 32;
    v16h a_next = a, b_next = b;
#pragma unroll
    for (int nt = 0; nt < 16; ++nt) {
      if (nt < 15) {
        b_next = load_frag(brow + (nt + 1) * 4096, kb, g);
      } else if (kt < 7) {
        a_next = load_frag(arow, kb + 32, g);
        b_next = load_frag(brow, kb + 32, g);
      }
      acc[nt] = wmma16(a, b, acc[nt]);
      b = b_next;
    }
    a = a_next;
  }

#pragma unroll
  for (int nt = 0; nt < 16; ++nt) {
    const int col = nt * 16 + ln;
    const float bias = bf[col];
#pragma unroll
    for (int r = 0; r < 8; ++r) {
      const int row = wave * 16 + g * 8 + r;
      out[(rowBase + row) * C_ + col] = acc[nt][r] + bias;
    }
  }
}

// ---------------------------------------------------------------------------
// Host launcher
// ---------------------------------------------------------------------------
extern "C" void kernel_launch(void* const* d_in, const int* in_sizes, int n_in,
                              void* d_out, int out_size, void* d_ws,
                              size_t ws_size, hipStream_t stream) {
  (void)in_sizes; (void)n_in; (void)out_size; (void)ws_size;
  const float* lig  = (const float*)d_in[0];
  const float* lnsc = (const float*)d_in[1];
  const float* lnbi = (const float*)d_in[2];
  const float* wq   = (const float*)d_in[3];
  const float* wk   = (const float*)d_in[4];
  const float* wv   = (const float*)d_in[5];
  const float* wf   = (const float*)d_in[6];
  const float* bf   = (const float*)d_in[7];
  float* out = (float*)d_out;

  char* ws = (char*)d_ws;
  size_t off = 0;
  const size_t big = (size_t)M_ * N_ * C_ * sizeof(half_t);  // 16 MB
  half_t* xh  = (half_t*)(ws + off); off += big;     // LN'd x (f16); reused as O
  half_t* wqT = (half_t*)(ws + off); off += 256 * 256 * sizeof(half_t);
  half_t* wkT = (half_t*)(ws + off); off += 256 * 256 * sizeof(half_t);
  half_t* wvT = (half_t*)(ws + off); off += 256 * 256 * sizeof(half_t);
  half_t* wfT = (half_t*)(ws + off); off += 256 * 256 * sizeof(half_t);
  half_t* Qw  = (half_t*)(ws + off); off += big;
  half_t* Kw  = (half_t*)(ws + off); off += big;
  half_t* Vw  = (half_t*)(ws + off); off += big;
  half_t* Ow  = xh;  // alias: xh dead after QKV projections

  const int SMEM_GEMM = 128 * 256 * 2 + 256 * 256 * 2;                    // 192 KB
  const int SMEM_ATTN = 2 * 256 * 32 * 2 + 32 * 256 * 2 + 256 * 256 * 4;  // 304 KB
  (void)hipFuncSetAttribute((const void*)qkv_kernel,
                            hipFuncAttributeMaxDynamicSharedMemorySize, SMEM_GEMM);
  (void)hipFuncSetAttribute((const void*)outproj_kernel,
                            hipFuncAttributeMaxDynamicSharedMemorySize, SMEM_GEMM);
  (void)hipFuncSetAttribute((const void*)attn_kernel,
                            hipFuncAttributeMaxDynamicSharedMemorySize, SMEM_ATTN);

  ln_kernel<<<dim3(M_ * N_ / 8), 256, 0, stream>>>(lig, lnsc, lnbi, xh);
  wcvt_kernel<<<dim3(256, 4), 256, 0, stream>>>(wq, wk, wv, wf, wqT, wkT, wvT, wfT);
  qkv_kernel<<<dim3(M_ * N_ / 128), 256, SMEM_GEMM, stream>>>(xh, wqT, Qw, 0);
  qkv_kernel<<<dim3(M_ * N_ / 128), 256, SMEM_GEMM, stream>>>(xh, wkT, Kw, 0);
  qkv_kernel<<<dim3(M_ * N_ / 128), 256, SMEM_GEMM, stream>>>(xh, wvT, Vw, 1);
  attn_kernel<<<dim3(M_, H_), 256, SMEM_ATTN, stream>>>(Qw, Kw, Vw, Ow);
  outproj_kernel<<<dim3(M_ * N_ / 128), 256, SMEM_GEMM, stream>>>(Ow, wfT, bf, out);
}
